// GAT_3624952398778
// MI455X (gfx1250) — matrix operand order repaired and checked by reading, not compiled, not run on hardware
//
#include <hip/hip_runtime.h>

#define N_NODES 50000
#define N_EDGES 800000
#define ETOT    (N_EDGES + N_NODES)
#define H_HEADS 4
#define C_CH    64
#define HC      256
#define IN_DIM  256
#define OUT_DIM 64
#define NEG_SLOPE 0.2f
#define BN_EPS  1e-5f

typedef __attribute__((ext_vector_type(16))) __bf16        v16bf;
typedef __attribute__((ext_vector_type(8)))  float         v8f;
typedef __attribute__((ext_vector_type(4)))  unsigned int  v4u;

union BFrag { v4u u[2]; v16bf v; };

// monotonic float <-> uint mapping for atomic max on floats
__device__ __forceinline__ unsigned fenc(float f) {
  unsigned u = __float_as_uint(f);
  return (u & 0x80000000u) ? ~u : (u | 0x80000000u);
}
__device__ __forceinline__ float fdec(unsigned k) {
  unsigned u = (k & 0x80000000u) ? (k ^ 0x80000000u) : ~k;
  return __uint_as_float(u);
}
#define MKEY_NEG_INF 0x007FFFFFu   // fenc(-inf)

// ---------------- prep kernels ----------------
__global__ void k_f32_to_bf16(const float* __restrict__ x, __bf16* __restrict__ y, int n) {
  int i = blockIdx.x * blockDim.x + threadIdx.x;
  if (i < n) y[i] = (__bf16)x[i];
}

// Wt[n*K + k] = (bf16) W[k*Nout + n]   (W is [K,Nout] row-major)
__global__ void k_transpose_bf16(const float* __restrict__ W, __bf16* __restrict__ Wt,
                                 int K, int Nout) {
  int i = blockIdx.x * blockDim.x + threadIdx.x;
  if (i >= K * Nout) return;
  int n = i / K, k = i - n * K;
  Wt[i] = (__bf16)W[k * Nout + n];
}

__global__ void k_build_edges(const long long* __restrict__ ei,
                              int* __restrict__ s, int* __restrict__ d) {
  int i = blockIdx.x * blockDim.x + threadIdx.x;
  if (i >= ETOT) return;
  if (i < N_EDGES) { s[i] = (int)ei[i]; d[i] = (int)ei[N_EDGES + i]; }
  else             { int n = i - N_EDGES; s[i] = n; d[i] = n; }
}

// ---------------- WMMA GEMM: C[M,Nout] = A[M,K](bf16) * Bt[Nout,K](bf16)^T + bias ----
__global__ __launch_bounds__(256)
void k_wmma_gemm(const __bf16* __restrict__ A, const __bf16* __restrict__ Bt,
                 const float* __restrict__ bias, float* __restrict__ Cm,
                 int M, int Nout, int K) {
  int wave = threadIdx.x >> 5;
  int lane = threadIdx.x & 31;
  int mt = blockIdx.x * 8 + wave;        // 16-row tile index
  if (mt * 16 >= M) return;
  int nbase = blockIdx.y * 64;

  int half = lane >> 4;                  // 0 / 1
  int lr   = lane & 15;

  // A fragment: lane = row M; 16 contiguous K at k + 16*half
  const __bf16* arow = A + (size_t)(mt * 16 + lr) * K + (half << 4);
  // B fragment: lane = col N; 16 contiguous K at k + 16*half
  const __bf16* b0 = Bt + (size_t)(nbase +  0 + lr) * K + (half << 4);
  const __bf16* b1 = Bt + (size_t)(nbase + 16 + lr) * K + (half << 4);
  const __bf16* b2 = Bt + (size_t)(nbase + 32 + lr) * K + (half << 4);
  const __bf16* b3 = Bt + (size_t)(nbase + 48 + lr) * K + (half << 4);

  v8f acc0 = {}, acc1 = {}, acc2 = {}, acc3 = {};
  for (int kb = 0; kb < K; kb += 32) {
    BFrag fa;
    fa.u[0] = *(const v4u*)(arow + kb);
    fa.u[1] = *(const v4u*)(arow + kb + 8);
    BFrag fb;
    fb.u[0] = *(const v4u*)(b0 + kb); fb.u[1] = *(const v4u*)(b0 + kb + 8);
    acc0 = __builtin_amdgcn_wmma_f32_16x16x32_bf16(false, fa.v, false, fb.v, (short)0, acc0, false, false);
    fb.u[0] = *(const v4u*)(b1 + kb); fb.u[1] = *(const v4u*)(b1 + kb + 8);
    acc1 = __builtin_amdgcn_wmma_f32_16x16x32_bf16(false, fa.v, false, fb.v, (short)0, acc1, false, false);
    fb.u[0] = *(const v4u*)(b2 + kb); fb.u[1] = *(const v4u*)(b2 + kb + 8);
    acc2 = __builtin_amdgcn_wmma_f32_16x16x32_bf16(false, fa.v, false, fb.v, (short)0, acc2, false, false);
    fb.u[0] = *(const v4u*)(b3 + kb); fb.u[1] = *(const v4u*)(b3 + kb + 8);
    acc3 = __builtin_amdgcn_wmma_f32_16x16x32_bf16(false, fa.v, false, fb.v, (short)0, acc3, false, false);
  }

  float bb0 = bias[nbase +  0 + lr];
  float bb1 = bias[nbase + 16 + lr];
  float bb2 = bias[nbase + 32 + lr];
  float bb3 = bias[nbase + 48 + lr];
  int rbase = mt * 16 + (half << 3);     // VGPR i -> row rbase+i (C/D layout)
#pragma unroll
  for (int i = 0; i < 8; ++i) {
    float* cp = Cm + (size_t)(rbase + i) * Nout;
    cp[nbase +  0 + lr] = acc0[i] + bb0;
    cp[nbase + 16 + lr] = acc1[i] + bb1;
    cp[nbase + 32 + lr] = acc2[i] + bb2;
    cp[nbase + 48 + lr] = acc3[i] + bb3;
  }
}

// ---------------- attention score dots ----------------
__global__ void k_attn_scores(const float* __restrict__ xh,
                              const float* __restrict__ att_src,
                              const float* __restrict__ att_dst,
                              float* __restrict__ asrc, float* __restrict__ adst) {
  int i = blockIdx.x * blockDim.x + threadIdx.x;   // (n,h)
  if (i >= N_NODES * H_HEADS) return;
  int n = i >> 2, h = i & 3;
  const float* xp = xh + (size_t)n * HC + h * C_CH;
  const float* as = att_src + h * C_CH;
  const float* ad = att_dst + h * C_CH;
  float s1 = 0.f, s2 = 0.f;
#pragma unroll 8
  for (int c = 0; c < C_CH; ++c) { float v = xp[c]; s1 += v * as[c]; s2 += v * ad[c]; }
  asrc[i] = s1; adst[i] = s2;
}

// ---------------- zero/init per layer ----------------
__global__ void k_init(float* __restrict__ agg, float* __restrict__ den,
                       unsigned* __restrict__ mkey, float* __restrict__ cs) {
  int i = blockIdx.x * blockDim.x + threadIdx.x;
  if (i < N_NODES * HC) agg[i] = 0.f;
  if (i < N_NODES * H_HEADS) { den[i] = 0.f; mkey[i] = MKEY_NEG_INF; }
  if (i < 2 * HC) cs[i] = 0.f;
}

// ---------------- per-edge softmax passes ----------------
__device__ __forceinline__ float edge_e(const float* asrc, const float* adst,
                                        int s, int d, int h) {
  float v = asrc[s * H_HEADS + h] + adst[d * H_HEADS + h];
  return v > 0.f ? v : NEG_SLOPE * v;
}

__global__ void k_edge_max(const int* __restrict__ s, const int* __restrict__ d,
                           const float* __restrict__ asrc, const float* __restrict__ adst,
                           unsigned* __restrict__ mkey) {
  int i = blockIdx.x * blockDim.x + threadIdx.x;
  if (i >= ETOT * H_HEADS) return;
  int e = i >> 2, h = i & 3;
  float v = edge_e(asrc, adst, s[e], d[e], h);
  atomicMax(&mkey[d[e] * H_HEADS + h], fenc(v));
}

__global__ void k_edge_den(const int* __restrict__ s, const int* __restrict__ d,
                           const float* __restrict__ asrc, const float* __restrict__ adst,
                           const unsigned* __restrict__ mkey, float* __restrict__ den) {
  int i = blockIdx.x * blockDim.x + threadIdx.x;
  if (i >= ETOT * H_HEADS) return;
  int e = i >> 2, h = i & 3;
  int dd = d[e];
  float v = edge_e(asrc, adst, s[e], dd, h);
  float m = fdec(mkey[dd * H_HEADS + h]);
  atomicAdd(&den[dd * H_HEADS + h], __expf(v - m));
}

// one 256-thread block per edge: scatter alpha * xh[src] into agg[dst]
__global__ __launch_bounds__(256)
void k_edge_agg(const int* __restrict__ s, const int* __restrict__ d,
                const float* __restrict__ asrc, const float* __restrict__ adst,
                const unsigned* __restrict__ mkey, const float* __restrict__ den,
                const float* __restrict__ xh, float* __restrict__ agg) {
  __shared__ float alpha[H_HEADS];
  int e = blockIdx.x;
  int src = s[e], dst = d[e];
  if (threadIdx.x < H_HEADS) {
    int h = threadIdx.x;
    float v = edge_e(asrc, adst, src, dst, h);
    float m = fdec(mkey[dst * H_HEADS + h]);
    float ex = __expf(v - m);
    alpha[h] = ex / (den[dst * H_HEADS + h] + 1e-16f);
  }
  __syncthreads();
  int t = threadIdx.x;
  float val = alpha[t >> 6] * xh[(size_t)src * HC + t];
  atomicAdd(&agg[(size_t)dst * HC + t], val);
}

// ---------------- batchnorm ----------------
__global__ __launch_bounds__(256)
void k_colstats(const float* __restrict__ x, float* __restrict__ cs) {
  int col = threadIdx.x;
  int r0 = blockIdx.x * 256;
  int r1 = r0 + 256; if (r1 > N_NODES) r1 = N_NODES;
  float s = 0.f, s2 = 0.f;
  for (int r = r0; r < r1; ++r) { float v = x[(size_t)r * HC + col]; s += v; s2 += v * v; }
  atomicAdd(&cs[col], s);
  atomicAdd(&cs[HC + col], s2);
}

__global__ void k_bn_bf16(const float* __restrict__ x, const float* __restrict__ cs,
                          const float* __restrict__ g, const float* __restrict__ beta,
                          __bf16* __restrict__ y, int n) {
  int i = blockIdx.x * blockDim.x + threadIdx.x;
  if (i >= n) return;
  int col = i & (HC - 1);
  const float inv_n = 1.f / (float)N_NODES;
  float mu  = cs[col] * inv_n;
  float var = cs[HC + col] * inv_n - mu * mu;
  float v = (x[i] - mu) * rsqrtf(var + BN_EPS) * g[col] + beta[col];
  y[i] = (__bf16)v;
}

// ---------------- log-softmax over 64 cols ----------------
__global__ __launch_bounds__(64)
void k_logsoftmax(const float* __restrict__ emb, float* __restrict__ out) {
  __shared__ float red[64];
  int r = blockIdx.x, t = threadIdx.x;
  float v = emb[(size_t)r * OUT_DIM + t];
  red[t] = v; __syncthreads();
  for (int st = 32; st > 0; st >>= 1) { if (t < st) red[t] = fmaxf(red[t], red[t + st]); __syncthreads(); }
  float m = red[0]; __syncthreads();
  red[t] = __expf(v - m); __syncthreads();
  for (int st = 32; st > 0; st >>= 1) { if (t < st) red[t] += red[t + st]; __syncthreads(); }
  float lse = m + __logf(red[0]);
  out[(size_t)r * OUT_DIM + t] = v - lse;
}

// ---------------- orchestration ----------------
extern "C" void kernel_launch(void* const* d_in, const int* in_sizes, int n_in,
                              void* d_out, int out_size, void* d_ws, size_t ws_size,
                              hipStream_t stream) {
  (void)in_sizes; (void)n_in; (void)out_size; (void)ws_size;
  const float*     x   = (const float*)d_in[0];
  const long long* ei  = (const long long*)d_in[1];
  const float* W1  = (const float*)d_in[2];
  const float* as1 = (const float*)d_in[3];
  const float* ad1 = (const float*)d_in[4];
  const float* b1  = (const float*)d_in[5];
  const float* W2  = (const float*)d_in[6];
  const float* as2 = (const float*)d_in[7];
  const float* ad2 = (const float*)d_in[8];
  const float* b2  = (const float*)d_in[9];
  const float* g1  = (const float*)d_in[10];
  const float* be1 = (const float*)d_in[11];
  const float* g2  = (const float*)d_in[12];
  const float* be2 = (const float*)d_in[13];
  const float* Wl  = (const float*)d_in[14];
  const float* bl  = (const float*)d_in[15];

  char* ws = (char*)d_ws; size_t off = 0;
  auto take = [&](size_t bytes) -> char* {
    char* p = ws + off;
    off = (off + bytes + 255) & ~(size_t)255;
    return p;
  };
  __bf16* xb   = (__bf16*)take((size_t)N_NODES * HC * 2);
  __bf16* Wt1  = (__bf16*)take((size_t)HC * HC * 2);
  __bf16* Wt2  = (__bf16*)take((size_t)HC * HC * 2);
  __bf16* Wtl  = (__bf16*)take((size_t)OUT_DIM * HC * 2);
  float*  xh   = (float*)take((size_t)N_NODES * HC * 4);
  float*  agg  = (float*)take((size_t)N_NODES * HC * 4);
  float*  asrc = (float*)take((size_t)N_NODES * H_HEADS * 4);
  float*  adst = (float*)take((size_t)N_NODES * H_HEADS * 4);
  int*    sI   = (int*)take((size_t)ETOT * 4);
  int*    dI   = (int*)take((size_t)ETOT * 4);
  unsigned* mkey = (unsigned*)take((size_t)N_NODES * H_HEADS * 4);
  float*  den  = (float*)take((size_t)N_NODES * H_HEADS * 4);
  float*  cs   = (float*)take((size_t)2 * HC * 4);

  float* outp = (float*)d_out;
  float* embp = outp + (size_t)N_NODES * OUT_DIM;

  const int TPB = 256;
  int nElem = N_NODES * HC;
  int nNH   = N_NODES * H_HEADS;
  int nEH   = ETOT * H_HEADS;

  // prep
  k_f32_to_bf16<<<(nElem + TPB - 1) / TPB, TPB, 0, stream>>>(x, xb, nElem);
  k_transpose_bf16<<<(HC * HC + TPB - 1) / TPB, TPB, 0, stream>>>(W1, Wt1, HC, HC);
  k_transpose_bf16<<<(HC * HC + TPB - 1) / TPB, TPB, 0, stream>>>(W2, Wt2, HC, HC);
  k_transpose_bf16<<<(HC * OUT_DIM + TPB - 1) / TPB, TPB, 0, stream>>>(Wl, Wtl, HC, OUT_DIM);
  k_build_edges<<<(ETOT + TPB - 1) / TPB, TPB, 0, stream>>>(ei, sI, dI);

  dim3 ggrid((N_NODES / 16 + 7) / 8, HC / 64);

  // ----- layer 1 -----
  k_wmma_gemm<<<ggrid, 256, 0, stream>>>(xb, Wt1, b1, xh, N_NODES, HC, IN_DIM);
  k_attn_scores<<<(nNH + TPB - 1) / TPB, TPB, 0, stream>>>(xh, as1, ad1, asrc, adst);
  k_init<<<(nElem + TPB - 1) / TPB, TPB, 0, stream>>>(agg, den, mkey, cs);
  k_edge_max<<<(nEH + TPB - 1) / TPB, TPB, 0, stream>>>(sI, dI, asrc, adst, mkey);
  k_edge_den<<<(nEH + TPB - 1) / TPB, TPB, 0, stream>>>(sI, dI, asrc, adst, mkey, den);
  k_edge_agg<<<ETOT, 256, 0, stream>>>(sI, dI, asrc, adst, mkey, den, xh, agg);
  k_colstats<<<(N_NODES + 255) / 256, 256, 0, stream>>>(agg, cs);
  k_bn_bf16<<<(nElem + TPB - 1) / TPB, TPB, 0, stream>>>(agg, cs, g1, be1, xb, nElem);

  // ----- layer 2 -----
  k_wmma_gemm<<<ggrid, 256, 0, stream>>>(xb, Wt2, b2, xh, N_NODES, HC, HC);
  k_attn_scores<<<(nNH + TPB - 1) / TPB, TPB, 0, stream>>>(xh, as2, ad2, asrc, adst);
  k_init<<<(nElem + TPB - 1) / TPB, TPB, 0, stream>>>(agg, den, mkey, cs);
  k_edge_max<<<(nEH + TPB - 1) / TPB, TPB, 0, stream>>>(sI, dI, asrc, adst, mkey);
  k_edge_den<<<(nEH + TPB - 1) / TPB, TPB, 0, stream>>>(sI, dI, asrc, adst, mkey, den);
  k_edge_agg<<<ETOT, 256, 0, stream>>>(sI, dI, asrc, adst, mkey, den, xh, agg);
  k_colstats<<<(N_NODES + 255) / 256, 256, 0, stream>>>(agg, cs);
  k_bn_bf16<<<(nElem + TPB - 1) / TPB, TPB, 0, stream>>>(agg, cs, g2, be2, xb, nElem);

  // ----- final linear + log-softmax -----
  dim3 fgrid((N_NODES / 16 + 7) / 8, OUT_DIM / 64);
  k_wmma_gemm<<<fgrid, 256, 0, stream>>>(xb, Wtl, bl, embp, N_NODES, OUT_DIM, HC);
  k_logsoftmax<<<N_NODES, 64, 0, stream>>>(embp, outp);
}
